// VisualDict_52424370815265
// MI455X (gfx1250) — matrix operand order repaired and checked by reading, not compiled, not run on hardware
//
#include <hip/hip_runtime.h>
#include <hip/hip_bf16.h>

// ---------------------------------------------------------------------------
// VisualDict (soft-VQ, top-5) for gfx1250.
//
//   k_cvt_x      : x f32 -> bf16 workspace
//   k_cvt_embed  : embed f32 -> bf16 + ||e||^2 per code
//   k_zero       : zero EMA accumulators
//   k_topk_wmma  : fused bf16 WMMA GEMM (x @ e^T) + running per-token top-5,
//                  codebook tiles streamed via GLOBAL_LOAD_ASYNC_TO_LDS_B128
//   k_scatter    : EMA stats via f32 global atomics (5-sparse)
//   k_cluster    : new_cluster_size + total reduction
//   k_new_embed  : normalized new codebook
//   k_quantize   : 5-way weighted gather -> quantize output
// ---------------------------------------------------------------------------

typedef __attribute__((ext_vector_type(16))) __bf16 v16bf;
typedef __attribute__((ext_vector_type(8)))  __bf16 v8bf;
typedef __attribute__((ext_vector_type(4)))  __bf16 v4bf;
typedef __attribute__((ext_vector_type(8)))  float  v8f;
typedef __attribute__((ext_vector_type(4)))  float  v4f;

static constexpr int N_TOK   = 65536;
static constexpr int K_CODES = 8192;
static constexpr int DIM     = 256;
static constexpr int TOPK_N  = 5;
static constexpr float DECAY_F = 0.1f;
static constexpr float EPS_F   = 1e-5f;

static constexpr int LDS_STRIDE = 264;   // 256 + 8 bf16 pad (16B) to spread banks

// --- gfx1250 async global->LDS helpers -------------------------------------
// Builtin (probe-confirmed arity/order): (global int4* src, lds int4* dst,
// imm offset, imm cpol). Pointee must be the 128-bit GCC vector type.
typedef int v4i_vs __attribute__((vector_size(16)));
#define AS1V(p) ((__attribute__((address_space(1))) v4i_vs*)(uintptr_t)(p))
#define AS3V(p) ((__attribute__((address_space(3))) v4i_vs*)(uintptr_t)(p))

__device__ __forceinline__ void async_copy_b128x2(const __bf16* g, __bf16* l) {
  // two 16B transfers per lane (offset applies to both global and LDS addr)
  __builtin_amdgcn_global_load_async_to_lds_b128(AS1V(g), AS3V(l), 0,  0);
  __builtin_amdgcn_global_load_async_to_lds_b128(AS1V(g), AS3V(l), 16, 0);
}

__device__ __forceinline__ void wait_async0() {
#if __has_builtin(__builtin_amdgcn_s_wait_asynccnt)
  __builtin_amdgcn_s_wait_asynccnt(0);
#else
  asm volatile("s_wait_asynccnt 0" ::: "memory");
#endif
}

// ---------------------------------------------------------------------------
__global__ __launch_bounds__(256) void k_cvt_x(const float* __restrict__ x,
                                               __bf16* __restrict__ xb) {
  int i = (blockIdx.x * 256 + threadIdx.x) * 4;
  v4f v = *(const v4f*)(x + i);
  v4bf o;
  o[0] = (__bf16)v[0]; o[1] = (__bf16)v[1];
  o[2] = (__bf16)v[2]; o[3] = (__bf16)v[3];
  *(v4bf*)(xb + i) = o;
}

__global__ __launch_bounds__(256) void k_cvt_embed(const float* __restrict__ embed,
                                                   __bf16* __restrict__ eb,
                                                   float* __restrict__ enorm) {
  __shared__ float red[256];
  const int row = blockIdx.x;
  const int tid = threadIdx.x;
  float v = embed[row * DIM + tid];
  eb[row * DIM + tid] = (__bf16)v;
  red[tid] = v * v;
  __syncthreads();
  for (int s = 128; s > 0; s >>= 1) {
    if (tid < s) red[tid] += red[tid + s];
    __syncthreads();
  }
  if (tid == 0) enorm[row] = red[0];
}

__global__ __launch_bounds__(256) void k_zero(float* __restrict__ p, int n4) {
  int i = blockIdx.x * 256 + threadIdx.x;
  if (i < n4) *(v4f*)(p + i * 4) = v4f{0.f, 0.f, 0.f, 0.f};
}

// ---------------------------------------------------------------------------
// Fused bf16 WMMA scores + running top-5 (see round-0 layout notes).
// score = x.e - 0.5*||e||^2 ; weights = softmax(2*score).
// ---------------------------------------------------------------------------
__global__ __launch_bounds__(256) void k_topk_wmma(
    const __bf16* __restrict__ xb, const __bf16* __restrict__ eb,
    const float* __restrict__ enorm,
    int* __restrict__ tk_idx, float* __restrict__ tk_w,
    float* __restrict__ out_idx) {
  __shared__ __align__(16) __bf16 sA[2][16 * LDS_STRIDE];
  __shared__ __align__(16) float s_en[K_CODES];

  const int tid  = threadIdx.x;
  const int lane = tid & 31;
  const int wave = tid >> 5;
  const int hi   = lane >> 4;          // 0 or 1 (which half of the wave)

  for (int i = tid; i < K_CODES; i += 256) s_en[i] = enorm[i];

  // Per-wave token tile: B operands held in registers for the whole loop.
  const int tok = (blockIdx.x * 8 + wave) * 16 + (lane & 15);
  v16bf B[8];
#pragma unroll
  for (int kk = 0; kk < 8; ++kk)
    B[kk] = *(const v16bf*)(xb + tok * DIM + kk * 32 + hi * 16);

  float ts[TOPK_N];
  int   ti[TOPK_N];
#pragma unroll
  for (int j = 0; j < TOPK_N; ++j) { ts[j] = -3.4e38f; ti[j] = 0; }

  // Stage code tile 0 into LDS buffer 0 (async copy, ASYNCcnt-tracked).
  const int srow = tid >> 4;
  const int scol = (tid & 15) * 16;
  async_copy_b128x2(eb + srow * DIM + scol, &sA[0][srow * LDS_STRIDE + scol]);
  wait_async0();
  __syncthreads();

  const int NT = K_CODES / 16;   // 512 code tiles
  const int arow = lane & 15;
  for (int t = 0; t < NT; ++t) {
    const int cur = t & 1;
    if (t + 1 < NT) {  // async-prefetch next code tile straight into LDS
      async_copy_b128x2(eb + ((t + 1) * 16 + srow) * DIM + scol,
                        &sA[cur ^ 1][srow * LDS_STRIDE + scol]);
    }

    // Preload all A operands, then issue the 8 WMMAs back-to-back.
    const __bf16* abase = &sA[cur][arow * LDS_STRIDE + hi * 16];
    v16bf A[8];
#pragma unroll
    for (int kk = 0; kk < 8; ++kk) {
      union { v16bf v; v8bf h[2]; } au;
      au.h[0] = *(const v8bf*)(abase + kk * 32);
      au.h[1] = *(const v8bf*)(abase + kk * 32 + 8);
      A[kk] = au.v;
    }
    v8f c = {};
#pragma unroll
    for (int kk = 0; kk < 8; ++kk)
      c = __builtin_amdgcn_wmma_f32_16x16x32_bf16(
              false, A[kk], false, B[kk], (short)0, c, false, false);

    // Running top-5 insert: guarded, branchless inner (v_cndmask chain).
    const int cbase = t * 16 + hi * 8;
    v4f en0 = *(const v4f*)&s_en[cbase];
    v4f en1 = *(const v4f*)&s_en[cbase + 4];
#pragma unroll
    for (int r = 0; r < 8; ++r) {
      const float en = (r < 4) ? en0[r] : en1[r - 4];
      float s = c[r] - 0.5f * en;
      if (s > ts[TOPK_N - 1]) {
        int idx = cbase + r;
#pragma unroll
        for (int j = 0; j < TOPK_N; ++j) {
          const bool  gt = s > ts[j];
          const float hs = gt ? s : ts[j];
          const float ls = gt ? ts[j] : s;
          const int   hidx = gt ? idx : ti[j];
          const int   lidx = gt ? ti[j] : idx;
          ts[j] = hs; ti[j] = hidx; s = ls; idx = lidx;
        }
      }
    }

    if (t + 1 < NT) wait_async0();  // own async loads landed in LDS
    __syncthreads();                // all waves' loads visible / reads done
  }

  // Merge the two lane halves (disjoint code sets for the same token).
  float os[TOPK_N]; int oi[TOPK_N];
#pragma unroll
  for (int j = 0; j < TOPK_N; ++j) {
    os[j] = __shfl_xor(ts[j], 16);
    oi[j] = __shfl_xor(ti[j], 16);
  }
#pragma unroll
  for (int j = 0; j < TOPK_N; ++j) {
    float s = os[j]; int idx = oi[j];
    if (s > ts[TOPK_N - 1]) {
#pragma unroll
      for (int jj = 0; jj < TOPK_N; ++jj) {
        const bool  gt = s > ts[jj];
        const float hs = gt ? s : ts[jj];
        const float ls = gt ? ts[jj] : s;
        const int   hidx = gt ? idx : ti[jj];
        const int   lidx = gt ? ti[jj] : idx;
        ts[jj] = hs; ti[jj] = hidx; s = ls; idx = lidx;
      }
    }
  }

  if (lane < 16) {
    // softmax(-dist) == softmax(2*score); shift by max for stability.
    const float m = ts[0];
    float w[TOPK_N], wsum = 0.f;
#pragma unroll
    for (int j = 0; j < TOPK_N; ++j) { w[j] = __expf(2.0f * (ts[j] - m)); wsum += w[j]; }
    const float inv = 1.0f / wsum;
#pragma unroll
    for (int j = 0; j < TOPK_N; ++j) {
      tk_idx[tok * TOPK_N + j] = ti[j];
      tk_w[tok * TOPK_N + j]   = w[j] * inv;
    }
    out_idx[tok] = (float)ti[0];   // encoding_indices (nearest code)
  }
}

// ---------------------------------------------------------------------------
__global__ __launch_bounds__(256) void k_scatter(
    const float* __restrict__ x, const int* __restrict__ tk_idx,
    const float* __restrict__ tk_w, float* __restrict__ enc_sum,
    float* __restrict__ emb_sum) {
  const int lane = threadIdx.x & 31;
  const int wave = threadIdx.x >> 5;
  const int t = blockIdx.x * 8 + wave;

  int il = 0; float wl = 0.f;
  if (lane < TOPK_N) {
    il = tk_idx[t * TOPK_N + lane];
    wl = tk_w[t * TOPK_N + lane];
  }
  v4f x0 = *(const v4f*)(x + t * DIM + lane * 8);
  v4f x1 = *(const v4f*)(x + t * DIM + lane * 8 + 4);
  if (lane < TOPK_N) atomicAdd(&enc_sum[il], wl);
#pragma unroll
  for (int j = 0; j < TOPK_N; ++j) {
    int   ij = __shfl(il, j);
    float wj = __shfl(wl, j);
    float* dst = emb_sum + ij * DIM + lane * 8;
#pragma unroll
    for (int r = 0; r < 4; ++r) atomicAdd(dst + r, wj * x0[r]);
#pragma unroll
    for (int r = 0; r < 4; ++r) atomicAdd(dst + 4 + r, wj * x1[r]);
  }
}

__global__ __launch_bounds__(256) void k_cluster(
    const float* __restrict__ cluster_size, const float* __restrict__ enc_sum,
    float* __restrict__ ncs, float* __restrict__ total) {
  __shared__ float red[256];
  const int k = blockIdx.x * 256 + threadIdx.x;
  float v = cluster_size[k] * DECAY_F + enc_sum[k] * (1.f - DECAY_F);
  ncs[k] = v;
  red[threadIdx.x] = v;
  __syncthreads();
  for (int s = 128; s > 0; s >>= 1) {
    if (threadIdx.x < s) red[threadIdx.x] += red[threadIdx.x + s];
    __syncthreads();
  }
  if (threadIdx.x == 0) atomicAdd(total, red[0]);
}

__global__ __launch_bounds__(256) void k_new_embed(
    const float* __restrict__ embed_avg, const float* __restrict__ emb_sum,
    const float* __restrict__ ncs, const float* __restrict__ total,
    float* __restrict__ new_embed) {
  const int i = blockIdx.x * 256 + threadIdx.x;   // < K*D
  const int k = i >> 8;
  const float T = *total;
  const float smoothed = (ncs[k] + EPS_F) / (T + (float)K_CODES * EPS_F);
  const float cs = smoothed * T;
  new_embed[i] = (embed_avg[i] * DECAY_F + emb_sum[i] * (1.f - DECAY_F)) / cs;
}

__global__ __launch_bounds__(256) void k_quantize(
    const float* __restrict__ new_embed, const int* __restrict__ tk_idx,
    const float* __restrict__ tk_w, float* __restrict__ out) {
  const int lane = threadIdx.x & 31;
  const int wave = threadIdx.x >> 5;
  const int t = blockIdx.x * 8 + wave;

  int il = 0; float wl = 0.f;
  if (lane < TOPK_N) {
    il = tk_idx[t * TOPK_N + lane];
    wl = tk_w[t * TOPK_N + lane];
  }
  float acc[8] = {0.f, 0.f, 0.f, 0.f, 0.f, 0.f, 0.f, 0.f};
#pragma unroll
  for (int j = 0; j < TOPK_N; ++j) {
    int   ij = __shfl(il, j);
    float wj = __shfl(wl, j);
    const float* src = new_embed + ij * DIM + lane * 8;
    v4f a = *(const v4f*)src;
    v4f b = *(const v4f*)(src + 4);
#pragma unroll
    for (int r = 0; r < 4; ++r) { acc[r] += wj * a[r]; acc[4 + r] += wj * b[r]; }
  }
  float* dst = out + t * DIM + lane * 8;
  *(v4f*)dst       = v4f{acc[0], acc[1], acc[2], acc[3]};
  *(v4f*)(dst + 4) = v4f{acc[4], acc[5], acc[6], acc[7]};
}

// ---------------------------------------------------------------------------
extern "C" void kernel_launch(void* const* d_in, const int* in_sizes, int n_in,
                              void* d_out, int out_size, void* d_ws, size_t ws_size,
                              hipStream_t stream) {
  const float* x            = (const float*)d_in[0];  // [N, D]
  const float* embed        = (const float*)d_in[1];  // [K, D]
  const float* cluster_size = (const float*)d_in[2];  // [K]
  // d_in[3] = cluster_sum (unused by the outputs)
  const float* embed_avg    = (const float*)d_in[4];  // [K, D]

  float* out = (float*)d_out;                 // quantize [N*D] then indices [N]

  // Workspace layout (bytes)
  char* ws = (char*)d_ws;
  size_t o = 0;
  __bf16* xb      = (__bf16*)(ws + o); o += (size_t)N_TOK * DIM * 2;
  __bf16* eb      = (__bf16*)(ws + o); o += (size_t)K_CODES * DIM * 2;
  float*  enorm   = (float*) (ws + o); o += (size_t)K_CODES * 4;
  int*    tk_idx  = (int*)   (ws + o); o += (size_t)N_TOK * TOPK_N * 4;
  float*  tk_w    = (float*) (ws + o); o += (size_t)N_TOK * TOPK_N * 4;
  float*  enc_sum = (float*) (ws + o); o += (size_t)K_CODES * 4;           // | zeroed
  float*  emb_sum = (float*) (ws + o); o += (size_t)K_CODES * DIM * 4;     // | region
  float*  total   = (float*) (ws + o); o += 256;                            // | (contig)
  float*  ncs     = (float*) (ws + o); o += (size_t)K_CODES * 4;
  float*  new_emb = (float*) (ws + o); o += (size_t)K_CODES * DIM * 4;

  // 1) precision conversions + code norms
  k_cvt_x<<<(N_TOK * DIM) / 1024, 256, 0, stream>>>(x, xb);
  k_cvt_embed<<<K_CODES, 256, 0, stream>>>(embed, eb, enorm);

  // 2) zero EMA accumulators (enc_sum + emb_sum + total are contiguous)
  const int zero_f = K_CODES + K_CODES * DIM + 64;
  const int zero4  = zero_f / 4;
  k_zero<<<(zero4 + 255) / 256, 256, 0, stream>>>(enc_sum, zero4);

  // 3) fused WMMA scores + top-5 + softmax
  k_topk_wmma<<<N_TOK / 128, 256, 0, stream>>>(xb, eb, enorm, tk_idx, tk_w,
                                               out + (size_t)N_TOK * DIM);

  // 4) EMA stats, codebook update, quantize
  k_scatter<<<N_TOK / 8, 256, 0, stream>>>(x, tk_idx, tk_w, enc_sum, emb_sum);
  k_cluster<<<K_CODES / 256, 256, 0, stream>>>(cluster_size, enc_sum, ncs, total);
  k_new_embed<<<(K_CODES * DIM) / 256, 256, 0, stream>>>(embed_avg, emb_sum, ncs,
                                                         total, new_emb);
  k_quantize<<<N_TOK / 8, 256, 0, stream>>>(new_emb, tk_idx, tk_w, out);
}
